// DLSTM_32882269618465
// MI455X (gfx1250) — compile-verified, matrix operand order checked by reading, not run on hardware
//
#include <hip/hip_runtime.h>
#include <cstdint>
#include <cstddef>

// Problem constants (match reference)
constexpr int Bb  = 32;    // batch
constexpr int Tt  = 1024;  // time steps
constexpr int Ii  = 512;   // input features
constexpr int Hh  = 512;   // hidden
constexpr int Gg  = 4;     // gates (i,f,c,o)
constexpr int NLV = 3;     // number of VLSTM layers
constexpr int SCAN_WGS = 16;          // persistent workgroups for the scan
constexpr int COLS = Hh / SCAN_WGS;   // 32 hidden columns owned per workgroup

typedef __attribute__((ext_vector_type(16))) _Float16 v16h;
typedef __attribute__((ext_vector_type(8)))  float    v8f;

union FragU { uint4 u[2]; v16h v; };

// ---------------------------------------------------------------------------
// CDNA5 async memory->LDS DMA (ASYNCcnt-tracked).  One 16B transfer per lane.
// Inline asm per CDNA5 ISA (portable across toolchains,
// unlike the arity-divergent TDM builtins).  Host pass falls back to a copy.
// ---------------------------------------------------------------------------
__device__ inline void async_copy_b128(void* lds_dst, const void* gsrc) {
#if defined(__AMDGCN__)
  unsigned loff =
      (unsigned)(uintptr_t)(__attribute__((address_space(3))) char*)lds_dst;
  unsigned long long gaddr = (unsigned long long)(uintptr_t)gsrc;
  asm volatile("global_load_async_to_lds_b128 %0, %1, off"
               :: "v"(loff), "v"(gaddr) : "memory");
#else
  *(uint4*)lds_dst = *(const uint4*)gsrc;
#endif
}

__device__ inline void wait_async_zero() {
#if defined(__AMDGCN__)
  asm volatile("s_wait_asynccnt 0x0" ::: "memory");
#endif
}

// Build a 16x32 (A, MxK) or 32x16 (B, KxN) f16 WMMA fragment from row-major data.
// Per ISA 7.12.2: lanes 0-15 hold row (lane), K = {k0..k0+7} U {k0+16..k0+23};
// lanes 16-31 hold row (lane-16), K = {k0+8..k0+15} U {k0+24..k0+31}.
__device__ inline v16h load_frag(const _Float16* __restrict__ base, int stride,
                                 int lane, int k0) {
  const _Float16* p = base + (size_t)(lane & 15) * stride + k0 + ((lane >> 4) << 3);
  FragU f;
  f.u[0] = *reinterpret_cast<const uint4*>(p);
  f.u[1] = *reinterpret_cast<const uint4*>(p + 16);
  return f.v;
}

// ---------------------------------------------------------------------------
// f32 -> f16 elementwise convert
// ---------------------------------------------------------------------------
__global__ void f32_to_f16_kernel(const float* __restrict__ src,
                                  _Float16* __restrict__ dst, size_t n) {
  for (size_t i = (size_t)blockIdx.x * blockDim.x + threadIdx.x; i < n;
       i += (size_t)gridDim.x * blockDim.x)
    dst[i] = (_Float16)src[i];
}

// src [G][K][N] f32  ->  dst [G][N][K] f16  (transpose so WMMA B-fragments are
// contiguous along K for each output column)
__global__ void transpose_cvt_kernel(const float* __restrict__ src,
                                     _Float16* __restrict__ dst,
                                     int G, int K, int N) {
  const size_t total = (size_t)G * K * N;
  for (size_t i = (size_t)blockIdx.x * blockDim.x + threadIdx.x; i < total;
       i += (size_t)gridDim.x * blockDim.x) {
    const size_t g   = i / ((size_t)K * N);
    const size_t rem = i - g * (size_t)K * N;
    const size_t n   = rem / K;
    const size_t k   = rem - n * K;
    dst[i] = (_Float16)src[(g * K + k) * N + n];
  }
}

// ---------------------------------------------------------------------------
// Pre-activation GEMM:  P[t,g,b,h] = Xa[b,t,:]·Wat[g][h,:]
//                                  (+ Xb[b,t,:]·Wbt[g][h,:])  + bias[g*gs + h]
// One block: one timestep t, one 16-row batch half, one gate; 8 waves, each
// wave owns 4 N-tiles.  K-outer / N-tile-inner loop keeps register pressure at
// 4 accumulators + 1 A-frag (no scratch spills); A comes from LDS (async-
// staged), B streams from L2-resident transposed weights.
// ---------------------------------------------------------------------------
__global__ void pregemm_kernel(const _Float16* __restrict__ Xa,   // [B][T][I]
                               const _Float16* __restrict__ Wat,  // [G][H][I]
                               const _Float16* __restrict__ Xb,   // [B][T][H] or null
                               const _Float16* __restrict__ Wbt,  // [G][H][H]
                               const float* __restrict__ bias,
                               int bias_gstride,
                               float* __restrict__ Pout)          // [T][G][B][H]
{
  extern __shared__ char smem[];
  _Float16* xLDS = (_Float16*)smem;        // 16 x I
  _Float16* hLDS = xLDS + 16 * Ii;         // 16 x H (only if Xb)
  const int tid  = threadIdx.x;
  const int lane = tid & 31;
  const int wave = tid >> 5;
  const int t    = blockIdx.x;
  const int b0   = blockIdx.y * 16;
  const int g    = blockIdx.z;

  // Async-stage the 16 activation rows for this (t, b-half) into LDS.
  {
    const int per_row = Ii / 8;  // uint4 per row
    for (int i = tid; i < 16 * per_row; i += blockDim.x) {
      const int row = i / per_row, off = i % per_row;
      async_copy_b128((uint4*)xLDS + i,
                      (const uint4*)(Xa + ((size_t)(b0 + row) * Tt + t) * Ii) + off);
    }
    if (Xb) {
      for (int i = tid; i < 16 * per_row; i += blockDim.x) {
        const int row = i / per_row, off = i % per_row;
        async_copy_b128((uint4*)hLDS + i,
                        (const uint4*)(Xb + ((size_t)(b0 + row) * Tt + t) * Hh) + off);
      }
    }
  }
  wait_async_zero();
  __syncthreads();

  constexpr int QT = 4;                 // N-tiles per wave (8 waves x 4 = 32)
  const int h0base = wave * QT * 16;
  v8f acc[QT];
#pragma unroll
  for (int q = 0; q < QT; ++q) acc[q] = {};

  const _Float16* waBase = Wat + ((size_t)g * Hh + h0base) * Ii;
  for (int k0 = 0; k0 < Ii; k0 += 32) {
    v16h a = load_frag(xLDS, Ii, lane, k0);
#pragma unroll
    for (int q = 0; q < QT; ++q) {
      v16h b = load_frag(waBase + (size_t)q * 16 * Ii, Ii, lane, k0);
      acc[q] = __builtin_amdgcn_wmma_f32_16x16x32_f16(false, a, false, b,
                                                      (short)0, acc[q], false, false);
    }
  }
  if (Xb) {
    const _Float16* wbBase = Wbt + ((size_t)g * Hh + h0base) * Hh;
    for (int k0 = 0; k0 < Hh; k0 += 32) {
      v16h a = load_frag(hLDS, Hh, lane, k0);
#pragma unroll
      for (int q = 0; q < QT; ++q) {
        v16h b = load_frag(wbBase + (size_t)q * 16 * Hh, Hh, lane, k0);
        acc[q] = __builtin_amdgcn_wmma_f32_16x16x32_f16(false, a, false, b,
                                                        (short)0, acc[q], false, false);
      }
    }
  }

  // D layout: VGPR r, lanes 0-15 -> M=r, N=lane; lanes 16-31 -> M=8+r, N=lane-16
  const int brow = (lane >> 4) << 3;
#pragma unroll
  for (int q = 0; q < QT; ++q) {
    const int h = h0base + q * 16 + (lane & 15);
    const float bvv = bias[g * bias_gstride + h];
    float* outp = Pout + (((size_t)t * Gg + g) * Bb) * Hh + h;
#pragma unroll
    for (int r = 0; r < 8; ++r)
      outp[(size_t)(b0 + brow + r) * Hh] = acc[q][r] + bvv;
  }
}

// ---------------------------------------------------------------------------
// Persistent recurrent scan.  16 cooperating workgroups; WG w owns hidden
// columns [w*32, w*32+32).  Recurrent weights for those columns live in LDS
// for the whole T loop (128 KB layer-1 / 32 KB VLSTM -- CDNA5 320KB WGP LDS).
// Cell state c stays in LDS; h ping-pongs through L2 (async-pulled into LDS
// each step).  The per-step rec GEMM is split into N-tile x M-tile x K-half
// tasks so all 8 waves stay busy even for the 32-column VLSTM case; the two
// K-half partial sums land in separate LDS planes summed during gate math.
// ---------------------------------------------------------------------------
__global__ void lstm_scan_kernel(const float* __restrict__ pre,   // [T][G][B][H]
                                 const _Float16* __restrict__ Ut, // [ngu][H][H] row-major [col][k]
                                 _Float16* __restrict__ hstate,   // [2][B][H] (zeroed)
                                 _Float16* __restrict__ hvec_h,   // [B][T][H] f16 (next layer input)
                                 float* __restrict__ out_hvec,    // [B][T][H] or null
                                 float* __restrict__ out_h,       // [B][H] or null
                                 float* __restrict__ out_c,       // [B][H] or null
                                 int* __restrict__ bar,           // zeroed counter
                                 int per_gate)                    // 1: 4 U mats, 0: shared u[0]
{
  extern __shared__ char smem[];
  const int ngu   = per_gate ? Gg : 1;
  const int ncols = ngu * COLS;                        // 128 or 32
  _Float16* uLDS = (_Float16*)smem;                    // ncols x 512
  _Float16* hLDS = uLDS + (size_t)ncols * Hh;          // 32 x 512
  float*  recLDS = (float*)(hLDS + (size_t)Bb * Hh);   // [2][Gg*COLS][Bb]
  float*  cLDS   = recLDS + 2 * Gg * COLS * Bb;        // 32 x 32
  constexpr int RECPLANE = Gg * COLS * Bb;

  const int tid    = threadIdx.x;
  const int lane   = tid & 31;
  const int wave   = tid >> 5;
  const int nwaves = blockDim.x >> 5;
  const int c0     = blockIdx.x * COLS;

  // Pin this WG's recurrent-weight slice (all gates) into LDS via async DMA.
  for (int g = 0; g < ngu; ++g) {
    const uint4* src = (const uint4*)(Ut + ((size_t)g * Hh + c0) * Hh);
    uint4* dst = (uint4*)(uLDS + (size_t)g * COLS * Hh);
    for (int i = tid; i < COLS * Hh / 8; i += blockDim.x)
      async_copy_b128(dst + i, src + i);
  }
  for (int i = tid; i < COLS * Bb; i += blockDim.x) cLDS[i] = 0.0f;
  wait_async_zero();
  __syncthreads();

  const int ntiles = ncols / 16;       // N tiles (8 or 2)
  const int ntasks = ntiles * 2 * 2;   // x 2 M-tiles x 2 K-halves (32 or 8)

  for (int t = 0; t < Tt; ++t) {
    // Async-pull the full h_{t-1} (32 KB f16) from L2 into LDS.
    {
      const uint4* src = (const uint4*)(hstate + (size_t)(t & 1) * Bb * Hh);
      uint4* dst = (uint4*)hLDS;
      for (int i = tid; i < Bb * Hh / 8; i += blockDim.x)
        async_copy_b128(dst + i, src + i);
    }
    if (t + 1 < Tt) {  // warm caches for the next step's preactivations
      const float* nxt = pre + (size_t)(t + 1) * Gg * Bb * Hh;
      __builtin_prefetch(nxt + (size_t)tid * 64, 0, 0);
    }
    wait_async_zero();
    __syncthreads();

    // rec = h_{t-1} @ U for this WG's columns (WMMA f16 -> f32, K split in 2)
    for (int task = wave; task < ntasks; task += nwaves) {
      const int kh = task & 1;
      const int mt = (task >> 1) & 1;
      const int nt = task >> 2;
      v8f acc = {};
      const _Float16* abase = hLDS + (size_t)mt * 16 * Hh;
      const _Float16* bbase = uLDS + (size_t)nt * 16 * Hh;
      const int kbeg = kh * (Hh / 2), kend = kbeg + (Hh / 2);
      for (int k0 = kbeg; k0 < kend; k0 += 32) {
        v16h a = load_frag(abase, Hh, lane, k0);
        v16h b = load_frag(bbase, Hh, lane, k0);
        acc = __builtin_amdgcn_wmma_f32_16x16x32_f16(false, a, false, b,
                                                     (short)0, acc, false, false);
      }
      const int col  = nt * 16 + (lane & 15);
      const int brow = mt * 16 + ((lane >> 4) << 3);
      float* plane = recLDS + kh * RECPLANE;
#pragma unroll
      for (int r = 0; r < 8; ++r) plane[col * Bb + brow + r] = acc[r];
    }
    __syncthreads();

    // Pointwise gate math (faithful to reference: additive c and h updates,
    // sigmoid on all four gates; VLSTM shares rec across gates).
    const float* pt = pre + (size_t)t * Gg * Bb * Hh;
    for (int j = tid; j < COLS * Bb; j += blockDim.x) {
      const int cc = j >> 5, b = j & 31;
      const int h  = c0 + cc;
      const int e0 = (per_gate ? (0 * COLS + cc) : cc) * Bb + b;
      const float r0 = recLDS[e0] + recLDS[RECPLANE + e0];
      float r1 = r0, r2 = r0, r3 = r0;
      if (per_gate) {
        const int e1 = (1 * COLS + cc) * Bb + b;
        const int e2 = (2 * COLS + cc) * Bb + b;
        const int e3 = (3 * COLS + cc) * Bb + b;
        r1 = recLDS[e1] + recLDS[RECPLANE + e1];
        r2 = recLDS[e2] + recLDS[RECPLANE + e2];
        r3 = recLDS[e3] + recLDS[RECPLANE + e3];
      }
      const float i_t = 1.0f / (1.0f + __expf(-(pt[(0 * Bb + b) * Hh + h] + r0)));
      const float f_t = 1.0f / (1.0f + __expf(-(pt[(1 * Bb + b) * Hh + h] + r1)));
      const float g_t = 1.0f / (1.0f + __expf(-(pt[(2 * Bb + b) * Hh + h] + r2)));
      const float o_t = 1.0f / (1.0f + __expf(-(pt[(3 * Bb + b) * Hh + h] + r3)));
      float cv = cLDS[j];
      cv = f_t + cv + i_t * g_t;
      cLDS[j] = cv;
      const float hv = o_t + tanhf(cv);
      hstate[(size_t)((t + 1) & 1) * Bb * Hh + (size_t)b * Hh + h] = (_Float16)hv;
      hvec_h[((size_t)b * Tt + t) * Hh + h] = (_Float16)hv;
      if (out_hvec) out_hvec[((size_t)b * Tt + t) * Hh + h] = hv;
      if (t == Tt - 1 && out_h) {
        out_h[(size_t)b * Hh + h] = hv;
        out_c[(size_t)b * Hh + h] = cv;
      }
    }

    // Device-wide step barrier (cluster fast-path is a NOP when not clustered).
    __threadfence();
    __builtin_amdgcn_s_cluster_barrier();
    __syncthreads();
    if (tid == 0) {
      __threadfence();
      atomicAdd(bar, 1);
      volatile int* vcnt = bar;
      const int target = (int)gridDim.x * (t + 1);
      while (*vcnt < target) __builtin_amdgcn_s_sleep(1);
      __threadfence();
    }
    __syncthreads();
  }
}

// ---------------------------------------------------------------------------
// Host-side orchestration
// ---------------------------------------------------------------------------
extern "C" void kernel_launch(void* const* d_in, const int* in_sizes, int n_in,
                              void* d_out, int out_size, void* d_ws, size_t ws_size,
                              hipStream_t stream) {
  (void)in_sizes; (void)n_in; (void)out_size; (void)ws_size;
  const float* x  = (const float*)d_in[0];  // [B][T][I]
  const float* w1 = (const float*)d_in[1];  // [4][I][H]
  const float* u1 = (const float*)d_in[2];  // [4][H][H]
  const float* b1 = (const float*)d_in[3];  // [4][H]
  const float* wx = (const float*)d_in[4];  // [L][4][I][H]
  const float* wh = (const float*)d_in[5];  // [L][4][H][H]
  const float* uv = (const float*)d_in[6];  // [L][4][H][H]  (only gate 0 used)
  const float* bv = (const float*)d_in[7];  // [L][4][H]     (only gate 0 used)

  float* out      = (float*)d_out;
  float* out_h    = out;                             // [B][H]
  float* out_c    = out + (size_t)Bb * Hh;           // [B][H]
  float* out_hvec = out + 2ull * Bb * Hh;            // [B][T][H]

  // Workspace carve-up
  char* ws = (char*)d_ws;
  auto alloc = [&](size_t bytes) -> char* {
    char* p = ws;
    ws += (bytes + 255) & ~size_t(255);
    return p;
  };
  _Float16* xh  = (_Float16*)alloc(sizeof(_Float16) * (size_t)Bb * Tt * Ii);
  _Float16* w1t = (_Float16*)alloc(sizeof(_Float16) * (size_t)Gg * Hh * Ii);
  _Float16* u1t = (_Float16*)alloc(sizeof(_Float16) * (size_t)Gg * Hh * Hh);
  _Float16* wxt = (_Float16*)alloc(sizeof(_Float16) * (size_t)NLV * Gg * Hh * Ii);
  _Float16* wht = (_Float16*)alloc(sizeof(_Float16) * (size_t)NLV * Gg * Hh * Hh);
  _Float16* uvt = (_Float16*)alloc(sizeof(_Float16) * (size_t)NLV * Hh * Hh);
  _Float16* hvh = (_Float16*)alloc(sizeof(_Float16) * (size_t)Bb * Tt * Hh);
  _Float16* hst = (_Float16*)alloc(sizeof(_Float16) * 2 * Bb * Hh);
  float*    pre = (float*)alloc(sizeof(float) * (size_t)Tt * Gg * Bb * Hh);
  int*      bar = (int*)alloc(256);

  // One-time precision conversion + weight transposes (B-fragments want [col][k]).
  f32_to_f16_kernel<<<2048, 256, 0, stream>>>(x, xh, (size_t)Bb * Tt * Ii);
  transpose_cvt_kernel<<<2048, 256, 0, stream>>>(w1, w1t, Gg, Ii, Hh);
  transpose_cvt_kernel<<<2048, 256, 0, stream>>>(u1, u1t, Gg, Hh, Hh);
  for (int l = 0; l < NLV; ++l) {
    transpose_cvt_kernel<<<2048, 256, 0, stream>>>(
        wx + (size_t)l * Gg * Ii * Hh, wxt + (size_t)l * Gg * Hh * Ii, Gg, Ii, Hh);
    transpose_cvt_kernel<<<2048, 256, 0, stream>>>(
        wh + (size_t)l * Gg * Hh * Hh, wht + (size_t)l * Gg * Hh * Hh, Gg, Hh, Hh);
    // VLSTM (source bug): all gates share u[0] -- transpose only gate 0.
    transpose_cvt_kernel<<<1024, 256, 0, stream>>>(
        uv + (size_t)l * Gg * Hh * Hh, uvt + (size_t)l * Hh * Hh, 1, Hh, Hh);
  }

  const dim3 pgrid(Tt, Bb / 16, Gg);
  const size_t pgLds1 = (size_t)16 * Ii * sizeof(_Float16);
  const size_t pgLds2 = (size_t)(16 * Ii + 16 * Hh) * sizeof(_Float16);
  const size_t scLdsCommon =
      (size_t)Bb * Hh * sizeof(_Float16) +
      (size_t)(2 * Gg * COLS * Bb + COLS * Bb) * sizeof(float);
  const size_t scLds1 = (size_t)Gg * COLS * Hh * sizeof(_Float16) + scLdsCommon; // ~196 KB
  const size_t scLdsV = (size_t)COLS * Hh * sizeof(_Float16) + scLdsCommon;      // ~100 KB

  // ---- Layer 1 (custom LSTM: per-gate bias, per-gate recurrent U) ----
  pregemm_kernel<<<pgrid, 256, pgLds1, stream>>>(xh, w1t, nullptr, nullptr, b1, Hh, pre);
  hipMemsetAsync(hst, 0, sizeof(_Float16) * 2 * Bb * Hh, stream);
  hipMemsetAsync(bar, 0, 256, stream);
  lstm_scan_kernel<<<SCAN_WGS, 256, scLds1, stream>>>(
      pre, u1t, hst, hvh, nullptr, nullptr, nullptr, bar, /*per_gate=*/1);

  // ---- VLSTM layers (x + previous layer h_vec; shared u[0], shared b[0]) ----
  for (int l = 0; l < NLV; ++l) {
    const bool last = (l == NLV - 1);
    pregemm_kernel<<<pgrid, 256, pgLds2, stream>>>(
        xh, wxt + (size_t)l * Gg * Hh * Ii,
        hvh, wht + (size_t)l * Gg * Hh * Hh,
        bv + (size_t)l * Gg * Hh, /*bias_gstride=*/0, pre);
    hipMemsetAsync(hst, 0, sizeof(_Float16) * 2 * Bb * Hh, stream);
    hipMemsetAsync(bar, 0, 256, stream);
    lstm_scan_kernel<<<SCAN_WGS, 256, scLdsV, stream>>>(
        pre, uvt + (size_t)l * Hh * Hh, hst, hvh,
        last ? out_hvec : nullptr, last ? out_h : nullptr,
        last ? out_c : nullptr, bar, /*per_gate=*/0);
  }
}
